// PDE_M1_55061480735237
// MI455X (gfx1250) — matrix operand order repaired and checked by reading, not compiled, not run on hardware
//
#include <hip/hip_runtime.h>

typedef __attribute__((ext_vector_type(16))) _Float16 v16h;
typedef __attribute__((ext_vector_type(8)))  _Float16 v8h;
typedef __attribute__((ext_vector_type(8)))  float    v8f;
typedef __attribute__((ext_vector_type(4)))  float    v4f;

#define N_MET 100000
#define N_RXN 500000
#define E_SUB 1000000
#define E_ALL 2000000
#define MSG   64
#define HID   128

static __device__ __forceinline__ v8f wmma16x16x32(v16h a, v16h b, v8f c) {
  // (neg_a, A, neg_b, B, c_mod, C, reuse_a, reuse_b)
  return __builtin_amdgcn_wmma_f32_16x16x32_f16(false, a, false, b, (short)0, c,
                                                false, false);
}

// Opaque LDS loads (asm volatile): cannot be hoisted/CSE'd out of the tile
// loop, so the compiler never spills these loop-invariant weights to scratch.
static __device__ __forceinline__ unsigned lds_off(const void* p) {
  return (unsigned)(unsigned long long)p;  // low 32 bits = LDS byte offset
}

static __device__ __forceinline__ v16h lds_load_frag(const _Float16* p) {
  v8h lo, hi;
  const unsigned off = lds_off(p);
  asm volatile("ds_load_b128 %0, %2\n\t"
               "ds_load_b128 %1, %2 offset:16\n\t"
               "s_wait_dscnt 0x0"
               : "=&v"(lo), "=&v"(hi)
               : "v"(off));
  union { v8h h[2]; v16h v; } u;
  u.h[0] = lo; u.h[1] = hi;
  return u.v;
}

// Load three 8-float runs (w0, w1, b) from LDS in one clause.
static __device__ __forceinline__ void lds_load_row8(const float* w0, const float* w1,
                                                     const float* b,
                                                     v4f& w0a, v4f& w0b,
                                                     v4f& w1a, v4f& w1b,
                                                     v4f& ba,  v4f& bb) {
  const unsigned o0 = lds_off(w0), o1 = lds_off(w1), o2 = lds_off(b);
  asm volatile("ds_load_b128 %0, %6\n\t"
               "ds_load_b128 %1, %6 offset:16\n\t"
               "ds_load_b128 %2, %7\n\t"
               "ds_load_b128 %3, %7 offset:16\n\t"
               "ds_load_b128 %4, %8\n\t"
               "ds_load_b128 %5, %8 offset:16\n\t"
               "s_wait_dscnt 0x0"
               : "=&v"(w0a), "=&v"(w0b), "=&v"(w1a), "=&v"(w1b), "=&v"(ba), "=&v"(bb)
               : "v"(o0), "v"(o1), "v"(o2));
}

// ---------------------------------------------------------------------------
// Init: zero h_rxn accumulator, seed out[m] = -lam*(conc - c_target)
// ---------------------------------------------------------------------------
__global__ void pde_init_kernel(const float* __restrict__ x,
                                const float* __restrict__ p,
                                const int*   __restrict__ met_type,
                                float* __restrict__ out,
                                float* __restrict__ h_rxn) {
  const int i      = blockIdx.x * blockDim.x + threadIdx.x;
  const int stride = gridDim.x * blockDim.x;
  for (int j = i; j < N_RXN * MSG; j += stride) h_rxn[j] = 0.0f;
  for (int m = i; m < N_MET; m += stride) {
    const int   ty  = met_type[m];
    const float lam = p[ty * 2 + 0];
    const float ct  = p[ty * 2 + 1];
    out[m] = -lam * (x[m * 8 + 3] - ct);
  }
}

// ---------------------------------------------------------------------------
// Edge MLP: per-wave 16-edge tiles.
//   h   = tanh([conc, sto] @ W1s + b1s)   (VALU; params via opaque LDS loads)
//   msg = h @ W2s + b2s                   (WMMA; W2s fragments in LDS)
//   atomicAdd into h_rxn[rxn_sub[e]*64 + j]
// ---------------------------------------------------------------------------
__global__ void pde_edge_mlp_kernel(const float* __restrict__ x,
                                    const float* __restrict__ sto_sub,
                                    const int*   __restrict__ met_sub,
                                    const int*   __restrict__ rxn_sub,
                                    const float* __restrict__ W1s,
                                    const float* __restrict__ b1s,
                                    const float* __restrict__ W2s,
                                    const float* __restrict__ b2s,
                                    float* __restrict__ h_rxn) {
  __shared__ alignas(32) _Float16 sB[16][32][16];  // 16 KB, fragment-ordered W2s
  __shared__ alignas(16) float sW0[HID];
  __shared__ alignas(16) float sW1c[HID];
  __shared__ alignas(16) float sb1[HID];
  __shared__ float sb2[MSG];

  for (int k = threadIdx.x; k < HID; k += blockDim.x) {
    sW0[k]  = W1s[k];
    sW1c[k] = W1s[HID + k];
    sb1[k]  = b1s[k];
  }
  for (int k = threadIdx.x; k < MSG; k += blockDim.x) sb2[k] = b2s[k];
  for (int idx = threadIdx.x; idx < 16 * 32; idx += blockDim.x) {
    const int frag = idx >> 5, l = idx & 31;
    const int kc = frag >> 2, nt = frag & 3;
    const int h2 = l >> 4, nn = l & 15;
#pragma unroll
    for (int j = 0; j < 16; ++j)
      sB[frag][l][j] = (_Float16)W2s[(kc * 32 + h2 * 8 + (j & 7) + (j >> 3) * 16) * MSG + nt * 16 + nn];
  }
  __syncthreads();

  const int lane   = threadIdx.x & 31;
  const int hi     = lane >> 4;
  const int n      = lane & 15;
  const int wpb    = blockDim.x >> 5;
  const int wid    = blockIdx.x * wpb + (threadIdx.x >> 5);
  const int nwaves = gridDim.x * wpb;

  const int ntiles = E_SUB / 16;
  for (int t = wid; t < ntiles; t += nwaves) {
    // Row data for A fragment: lane covers row m = n (both half-waves).
    const int   e    = t * 16 + n;
    const float conc = x[met_sub[e] * 8 + 3];
    const float sto  = sto_sub[e];

    v8f C[4];
#pragma unroll
    for (int nt = 0; nt < 4; ++nt) {
      const float bias = sb2[nt * 16 + n];
#pragma unroll
      for (int v = 0; v < 8; ++v) C[nt][v] = bias;
    }

#pragma unroll
    for (int kc = 0; kc < 4; ++kc) {
      // A(16x32) layout: halves 0..7 -> K = hi*8+j, halves 8..15 -> K+16.
      v16h A;
      {
        const int k0 = kc * 32 + hi * 8;
        v4f w0a, w0b, w1a, w1b, ba, bb;
        lds_load_row8(&sW0[k0], &sW1c[k0], &sb1[k0], w0a, w0b, w1a, w1b, ba, bb);
#pragma unroll
        for (int j = 0; j < 4; ++j) {
          A[j]     = (_Float16)tanhf(fmaf(conc, w0a[j], fmaf(sto, w1a[j], ba[j])));
          A[4 + j] = (_Float16)tanhf(fmaf(conc, w0b[j], fmaf(sto, w1b[j], bb[j])));
        }
        const int k1 = k0 + 16;
        lds_load_row8(&sW0[k1], &sW1c[k1], &sb1[k1], w0a, w0b, w1a, w1b, ba, bb);
#pragma unroll
        for (int j = 0; j < 4; ++j) {
          A[8 + j]  = (_Float16)tanhf(fmaf(conc, w0a[j], fmaf(sto, w1a[j], ba[j])));
          A[12 + j] = (_Float16)tanhf(fmaf(conc, w0b[j], fmaf(sto, w1b[j], bb[j])));
        }
      }
#pragma unroll
      for (int nt = 0; nt < 4; ++nt) {
        const v16h B = lds_load_frag(&sB[kc * 4 + nt][lane][0]);
        C[nt] = wmma16x16x32(A, B, C[nt]);
      }
    }

    // C/D layout: lane owns rows m = v + 8*hi, column n.
    int rdst[8];
#pragma unroll
    for (int v = 0; v < 8; ++v) rdst[v] = rxn_sub[t * 16 + v + 8 * hi];
#pragma unroll
    for (int nt = 0; nt < 4; ++nt)
#pragma unroll
      for (int v = 0; v < 8; ++v)
        atomicAdd(&h_rxn[rdst[v] * MSG + nt * 16 + n], C[nt][v]);
  }
}

// ---------------------------------------------------------------------------
// Reaction MLP: per-wave 16-reaction tiles.
//   rate = tanh(h_rxn @ W1n + b1n) @ W2n + b2n   (WMMA + shuffle reduce)
//   v    = 10^log_k * rate
// ---------------------------------------------------------------------------
__global__ void pde_rxn_mlp_kernel(const float* __restrict__ h_rxn,
                                   const float* __restrict__ W1n,
                                   const float* __restrict__ b1n,
                                   const float* __restrict__ W2n,
                                   const float* __restrict__ b2n,
                                   const float* __restrict__ log_k,
                                   float* __restrict__ vout) {
  __shared__ alignas(32) _Float16 sB[16][32][16];  // 16 KB, fragment-ordered W1n
  __shared__ float sb1n[HID], sW2n[HID];

  for (int k = threadIdx.x; k < HID; k += blockDim.x) {
    sb1n[k] = b1n[k];
    sW2n[k] = W2n[k];
  }
  for (int idx = threadIdx.x; idx < 16 * 32; idx += blockDim.x) {
    const int frag = idx >> 5, l = idx & 31;
    const int kc = frag >> 3, nt = frag & 7;
    const int h2 = l >> 4, nn = l & 15;
#pragma unroll
    for (int j = 0; j < 16; ++j)
      sB[frag][l][j] = (_Float16)W1n[(kc * 32 + h2 * 8 + (j & 7) + (j >> 3) * 16) * HID + nt * 16 + nn];
  }
  __syncthreads();

  const int lane   = threadIdx.x & 31;
  const int hi     = lane >> 4;
  const int n      = lane & 15;
  const int wpb    = blockDim.x >> 5;
  const int wid    = blockIdx.x * wpb + (threadIdx.x >> 5);
  const int nwaves = gridDim.x * wpb;

  const float b2 = b2n[0];
  const int ntiles = N_RXN / 16;
  for (int t = wid; t < ntiles; t += nwaves) {
    // A fragments from h_rxn (f32 -> f16), row m = n.
    v16h A[2];
    const int base = (t * 16 + n) * MSG;
#pragma unroll
    for (int kc = 0; kc < 2; ++kc)
#pragma unroll
      for (int j = 0; j < 8; ++j) {
        A[kc][j]     = (_Float16)h_rxn[base + kc * 32 + hi * 8 + j];
        A[kc][8 + j] = (_Float16)h_rxn[base + kc * 32 + 16 + hi * 8 + j];
      }

    float s[8];
#pragma unroll
    for (int v = 0; v < 8; ++v) s[v] = 0.0f;

#pragma unroll
    for (int nt = 0; nt < 8; ++nt) {
      v8f C;
      const float bias = sb1n[nt * 16 + n];
#pragma unroll
      for (int v = 0; v < 8; ++v) C[v] = bias;
      {
        const v16h B0 = lds_load_frag(&sB[0 * 8 + nt][lane][0]);
        C = wmma16x16x32(A[0], B0, C);
        const v16h B1 = lds_load_frag(&sB[1 * 8 + nt][lane][0]);
        C = wmma16x16x32(A[1], B1, C);
      }
      const float w2 = sW2n[nt * 16 + n];
#pragma unroll
      for (int v = 0; v < 8; ++v) s[v] += tanhf(C[v]) * w2;
    }

    // Reduce over the 16 lanes of each half-wave (columns n = 0..15).
#pragma unroll
    for (int off = 1; off < 16; off <<= 1)
#pragma unroll
      for (int v = 0; v < 8; ++v) s[v] += __shfl_xor(s[v], off, 32);

    if (n == 0) {
#pragma unroll
      for (int v = 0; v < 8; ++v) {
        const int r = t * 16 + v + 8 * hi;
        vout[r] = exp10f(log_k[r]) * (s[v] + b2);
      }
    }
  }
}

// ---------------------------------------------------------------------------
// Final scatter: dxdt[met_all[e]] += sto_all[e] * v[rxn_all[e]]
// ---------------------------------------------------------------------------
__global__ void pde_scatter_kernel(const float* __restrict__ sto_all,
                                   const int*   __restrict__ rxn_all,
                                   const int*   __restrict__ met_all,
                                   const float* __restrict__ vout,
                                   float* __restrict__ out) {
  const int i      = blockIdx.x * blockDim.x + threadIdx.x;
  const int stride = gridDim.x * blockDim.x;
  for (int e = i; e < E_ALL; e += stride)
    atomicAdd(&out[met_all[e]], sto_all[e] * vout[rxn_all[e]]);
}

// ---------------------------------------------------------------------------
extern "C" void kernel_launch(void* const* d_in, const int* in_sizes, int n_in,
                              void* d_out, int out_size, void* d_ws, size_t ws_size,
                              hipStream_t stream) {
  const float* x        = (const float*)d_in[0];
  const float* sto_sub  = (const float*)d_in[1];
  const float* sto_all  = (const float*)d_in[2];
  const float* log_k    = (const float*)d_in[3];
  const float* W1s      = (const float*)d_in[4];
  const float* b1s      = (const float*)d_in[5];
  const float* W2s      = (const float*)d_in[6];
  const float* b2s      = (const float*)d_in[7];
  const float* W1n      = (const float*)d_in[8];
  const float* b1n      = (const float*)d_in[9];
  const float* W2n      = (const float*)d_in[10];
  const float* b2n      = (const float*)d_in[11];
  const float* p        = (const float*)d_in[12];
  const int*   met_sub  = (const int*)d_in[13];
  const int*   rxn_sub  = (const int*)d_in[14];
  const int*   met_all  = (const int*)d_in[15];
  const int*   rxn_all  = (const int*)d_in[16];
  const int*   met_type = (const int*)d_in[17];
  float*       out      = (float*)d_out;

  float* h_rxn = (float*)d_ws;                       // N_RXN*MSG f32 (128 MB)
  float* vout  = h_rxn + (size_t)N_RXN * MSG;        // N_RXN f32 (2 MB)

  pde_init_kernel<<<4096, 256, 0, stream>>>(x, p, met_type, out, h_rxn);
  pde_edge_mlp_kernel<<<2048, 256, 0, stream>>>(x, sto_sub, met_sub, rxn_sub,
                                                W1s, b1s, W2s, b2s, h_rxn);
  pde_rxn_mlp_kernel<<<1024, 256, 0, stream>>>(h_rxn, W1n, b1n, W2n, b2n,
                                               log_k, vout);
  pde_scatter_kernel<<<2048, 256, 0, stream>>>(sto_all, rxn_all, met_all, vout, out);
}
